// LSTMCell_72962904424866
// MI455X (gfx1250) — matrix operand order repaired and checked by reading, not compiled, not run on hardware
//
#include <hip/hip_runtime.h>

// ---------------------------------------------------------------------------
// LSTM forward for MI455X (gfx1250), wave32 + WMMA bf16 16x16x32.
// Batch-parallel recurrence: each workgroup owns 16 batch rows for all T
// steps; h/c live in LDS. 8 waves x 4 N-tiles (512-VGPR/wave budget at 2
// waves/SIMD): 48 of 52 B fragments register-resident for all 270 steps
// (384 VGPRs); one K-tile (4 fragments) streamed per step via volatile asm
// global_load_b128, consumed 8 K-tiles later. Zero spills by construction.
// ---------------------------------------------------------------------------

static constexpr int kH   = 128;          // hidden
static constexpr int kI   = 270;          // input features
static constexpr int kIP  = 288;          // input features padded to 32
static constexpr int kC   = 3;            // classes
static constexpr int kB   = 1024;         // batch
static constexpr int kT   = 270;          // timesteps
static constexpr int kBM  = 16;           // batch rows per workgroup (= WMMA M)
static constexpr int kGN  = 4 * kH;       // 512 gate outputs (f,i,c,o)
static constexpr int kNT  = kGN / 16;     // 32 N-tiles
static constexpr int kKTX = kIP / 32;     // 9 K-tiles, input projection
static constexpr int kKTH = kH / 32;      // 4 K-tiles, recurrent
static constexpr int kKTT = kKTX + kKTH;  // 13 unified K-tiles
static constexpr int kStreamKt = 8;       // the K-tile streamed from L2
static constexpr int kRow = kI + kH;      // 398, row length of Wf/Wi/Wc/Wo
static constexpr int kThreads = 256;      // 8 wave32s

typedef __attribute__((ext_vector_type(16))) __bf16 v16bf;
typedef __attribute__((ext_vector_type(8)))  float  v8f;

__device__ __forceinline__ unsigned short f2bf(float f) {
  union { float f; unsigned u; } v; v.f = f;
  unsigned r = v.u + 0x7FFFu + ((v.u >> 16) & 1u);   // round-to-nearest-even
  return (unsigned short)(r >> 16);
}

// A-fragment: 16-bit A 16x32 layout (ISA 7.12.2). Lane L: row m = L&15,
// half = L>>4; elems 0..7 -> K = k0..k0+7, elems 8..15 -> K = k0+16..k0+23,
// k0 = kt*32 + half*8. Two 16B chunks from row-major LDS.
__device__ __forceinline__ v16bf load_a(const unsigned short* p) {
  union { v16bf v; uint4 q[2]; } u;
  u.q[0] = *(const uint4*)(p);
  u.q[1] = *(const uint4*)(p + 16);
  return u.v;
}

// B-fragment: packed contiguously per (tile,lane) by the prep kernel.
__device__ __forceinline__ v16bf load_b(const unsigned short* p) {
  union { v16bf v; uint4 q[2]; } u;
  u.q[0] = ((const uint4*)p)[0];
  u.q[1] = ((const uint4*)p)[1];
  return u.v;
}

__device__ __forceinline__ v16bf pack_b(uint4 lo, uint4 hi) {
  union { v16bf v; uint4 q[2]; } u;
  u.q[0] = lo; u.q[1] = hi;
  return u.v;
}

// ---------------------------------------------------------------------------
// Repack gate weights into bf16 WMMA B-tiles.
//  Bx[k=i][n=g*128+j]  = Wg[j][128 + i]   (K padded 270->288 with zeros)
//  Bh[k=h][n=g*128+j]  = Wg[j][h]
// Tile element addressing: lane L, elem e -> k = kt*32+(L>>4)*8+(e&7)+(e>>3)*16,
// n = nt*16 + (L&15)   (16-bit B 32x16 layout).
// ---------------------------------------------------------------------------
__global__ void prep_weights(const float* __restrict__ Wf,
                             const float* __restrict__ Wi,
                             const float* __restrict__ Wc,
                             const float* __restrict__ Wo,
                             unsigned short* __restrict__ BX,
                             unsigned short* __restrict__ BH) {
  int idx = blockIdx.x * blockDim.x + threadIdx.x;
  const float* W[4] = {Wf, Wi, Wc, Wo};
  const int nbx = kKTX * kNT * 512;
  const int nbh = kKTH * kNT * 512;
  if (idx < nbx) {
    int e    = idx & 15;
    int lane = (idx >> 4) & 31;
    int tile = idx >> 9;
    int nt   = tile % kNT;
    int kt   = tile / kNT;
    int k = kt * 32 + (lane >> 4) * 8 + (e & 7) + (e >> 3) * 16;
    int n = nt * 16 + (lane & 15);
    int g = n >> 7, j = n & 127;
    float v = (k < kI) ? W[g][j * kRow + kH + k] : 0.0f;
    BX[idx] = f2bf(v);
  } else if ((idx -= nbx) < nbh) {
    int e    = idx & 15;
    int lane = (idx >> 4) & 31;
    int tile = idx >> 9;
    int nt   = tile % kNT;
    int kt   = tile / kNT;
    int k = kt * 32 + (lane >> 4) * 8 + (e & 7) + (e >> 3) * 16;
    int n = nt * 16 + (lane & 15);
    int g = n >> 7, j = n & 127;
    BH[idx] = f2bf(W[g][j * kRow + k]);
  }
}

// ---------------------------------------------------------------------------
// Persistent per-batch-slice LSTM. grid = B/16 workgroups, 256 threads.
// ---------------------------------------------------------------------------
__global__ __launch_bounds__(kThreads)
void lstm_persistent(const float* __restrict__ x,
                     const float* __restrict__ Wy,
                     const float* __restrict__ bfv, const float* __restrict__ biv,
                     const float* __restrict__ bcv, const float* __restrict__ bov,
                     const float* __restrict__ byv,
                     const unsigned short* __restrict__ BX,
                     const unsigned short* __restrict__ BH,
                     float* __restrict__ outH, float* __restrict__ outP) {
  __shared__ __align__(16) unsigned short x_s[kBM * kIP];   //  9216 B
  __shared__ __align__(16) unsigned short h_bf[kBM * kH];   //  4096 B
  __shared__ float pre_s[kBM * kGN];                        // 32768 B (reused as
                                                            //  softmax scratch)
  __shared__ float c_s[kBM * kH];                           //  8192 B
  __shared__ __align__(16) float b_s[4 * kH];               //  2048 B biases
  __shared__ float wy_s[kC * kH];                           //  1536 B
  __shared__ float by_s[4];                                 //    16 B
                                                            // total 57,872 B

  const int tid  = threadIdx.x;
  const int lane = tid & 31;
  const int wave = tid >> 5;                // 0..7
  const int b0   = blockIdx.x * kBM;

  // --- init: h = c = 0, stage biases / Wy / by ---
  for (int i = tid; i < kBM * kH; i += kThreads) { c_s[i] = 0.f; h_bf[i] = 0; }
  for (int i = tid; i < kH; i += kThreads) {
    b_s[0 * kH + i] = bfv[i];
    b_s[1 * kH + i] = biv[i];
    b_s[2 * kH + i] = bcv[i];
    b_s[3 * kH + i] = bov[i];
  }
  for (int i = tid; i < kC * kH; i += kThreads) wy_s[i] = Wy[i];
  if (tid < kC) by_s[tid] = byv[tid];

  // gate epilogue ownership: row mrow = tid>>4, cols jb..jb+7 (16 thr/row)
  const int mrow = tid >> 4;
  const int jb   = (tid & 15) * 8;

  // --- weight-stationary: 48 of 52 B fragments resident in registers ---
  const int ntBase = wave * 4;              // 8 waves x 4 N-tiles = 32
  v16bf br[(kKTT - 1) * 4];
#pragma unroll
  for (int ktt = 0; ktt < kKTT; ++ktt) {
    if (ktt == kStreamKt) continue;
    int r = (ktt < kStreamKt) ? ktt : ktt - 1;
#pragma unroll
    for (int q = 0; q < 4; ++q) {
      const unsigned short* p = (ktt < kKTX)
          ? &BX[((ktt * kNT + ntBase + q) * 32 + lane) * 16]
          : &BH[(((ktt - kKTX) * kNT + ntBase + q) * 32 + lane) * 16];
      br[r * 4 + q] = load_b(p);
    }
  }
  // per-step streamed tile (x K-tile 8), four N-tiles, 1024 B apart
  const unsigned short* spB =
      &BX[((kStreamKt * kNT + ntBase) * 32 + lane) * 16];

  __syncthreads();

  const int am    = lane & 15;
  const int ahalf = lane >> 4;

  for (int t = 0; t < kT; ++t) {
    // --- stage x_t (fp32 -> bf16, pad 270->288); prefetch next timestep ---
    for (int idx = tid; idx < kBM * kIP; idx += kThreads) {
      int m = idx / kIP, i = idx - m * kIP;
      float v = 0.0f;
      if (i < kI) {
        const float* px = &x[((size_t)(b0 + m) * kT + t) * kI + i];
        v = *px;
        if (t + 1 < kT) __builtin_prefetch(px + kI, 0, 1);  // global_prefetch_b8
      }
      x_s[idx] = f2bf(v);
    }
    __syncthreads();

    // --- issue streamed B-tile loads (volatile: never hoisted/spilled) ---
    uint4 s0a, s0b, s1a, s1b, s2a, s2b, s3a, s3b;
    asm volatile("global_load_b128 %0, %8, off\n\t"
                 "global_load_b128 %1, %8, off offset:16\n\t"
                 "global_load_b128 %2, %8, off offset:1024\n\t"
                 "global_load_b128 %3, %8, off offset:1040\n\t"
                 "global_load_b128 %4, %8, off offset:2048\n\t"
                 "global_load_b128 %5, %8, off offset:2064\n\t"
                 "global_load_b128 %6, %8, off offset:3072\n\t"
                 "global_load_b128 %7, %8, off offset:3088"
                 : "=&v"(s0a), "=&v"(s0b), "=&v"(s1a), "=&v"(s1b),
                   "=&v"(s2a), "=&v"(s2b), "=&v"(s3a), "=&v"(s3b)
                 : "v"((unsigned long long)(uintptr_t)spB)
                 : "memory");

    // --- pre[16,512] = x_t @ Bx + h @ Bh via WMMA bf16 ---
    v8f acc[4] = {};
#pragma unroll
    for (int ktt = 0; ktt < kKTT; ++ktt) {
      const unsigned short* ap = (ktt < kKTX)
          ? &x_s[am * kIP + ktt * 32 + ahalf * 8]
          : &h_bf[am * kH + (ktt - kKTX) * 32 + ahalf * 8];
      v16bf a = load_a(ap);
      if (ktt == kStreamKt) {
        asm volatile("s_wait_loadcnt 0x0" ::: "memory");
        acc[0] = __builtin_amdgcn_wmma_f32_16x16x32_bf16(
            false, a, false, pack_b(s0a, s0b), (short)0, acc[0], false, false);
        acc[1] = __builtin_amdgcn_wmma_f32_16x16x32_bf16(
            false, a, false, pack_b(s1a, s1b), (short)0, acc[1], false, false);
        acc[2] = __builtin_amdgcn_wmma_f32_16x16x32_bf16(
            false, a, false, pack_b(s2a, s2b), (short)0, acc[2], false, false);
        acc[3] = __builtin_amdgcn_wmma_f32_16x16x32_bf16(
            false, a, false, pack_b(s3a, s3b), (short)0, acc[3], false, false);
      } else {
        int r = (ktt < kStreamKt) ? ktt : ktt - 1;
#pragma unroll
        for (int q = 0; q < 4; ++q)
          acc[q] = __builtin_amdgcn_wmma_f32_16x16x32_bf16(
              false, a, false, br[r * 4 + q], (short)0, acc[q], false, false);
      }
    }
    // C/D layout: lane L holds col n=L&15; VGPR vg holds row (L>>4)*8 + vg.
    {
      int mr = (lane >> 4) * 8;
#pragma unroll
      for (int q = 0; q < 4; ++q) {
        int n = (ntBase + q) * 16 + (lane & 15);
#pragma unroll
        for (int vg = 0; vg < 8; ++vg) pre_s[(mr + vg) * kGN + n] = acc[q][vg];
      }
    }
    __syncthreads();

    // --- gates + cell update (fp32), write hiddens (biases from LDS) ---
#pragma unroll
    for (int u = 0; u < 8; ++u) {
      int j = jb + u;
      float pf = pre_s[mrow * kGN + 0 * kH + j] + b_s[0 * kH + j];
      float pi = pre_s[mrow * kGN + 1 * kH + j] + b_s[1 * kH + j];
      float pg = pre_s[mrow * kGN + 2 * kH + j] + b_s[2 * kH + j];
      float po = pre_s[mrow * kGN + 3 * kH + j] + b_s[3 * kH + j];
      float fg = 1.0f / (1.0f + __expf(-pf));
      float ig = 1.0f / (1.0f + __expf(-pi));
      float gg = tanhf(pg);
      float og = 1.0f / (1.0f + __expf(-po));
      float cn = fg * c_s[mrow * kH + j] + ig * gg;
      float hn = og * tanhf(cn);
      c_s[mrow * kH + j]  = cn;
      h_bf[mrow * kH + j] = f2bf(hn);
      outH[((size_t)(b0 + mrow) * kT + t) * kH + j] = hn;
    }
    __syncthreads();

    // --- logits (C=3): parallel partial dots into pre_s (now free) ---
    if (tid < kBM * kC * 4) {               // 192 threads
      int m  = tid / 12;
      int r  = tid - m * 12;
      int cc = r >> 2;
      int s  = r & 3;                       // 32-element slice of H
      const uint4* hp = (const uint4*)&h_bf[m * kH + s * 32];
      const float* wp = &wy_s[cc * kH + s * 32];
      float p = 0.f;
#pragma unroll
      for (int v2 = 0; v2 < 4; ++v2) {
        uint4 qq = hp[v2];
        unsigned uu[4] = {qq.x, qq.y, qq.z, qq.w};
#pragma unroll
        for (int u = 0; u < 4; ++u) {
          float lo = __uint_as_float(uu[u] << 16);
          float hi = __uint_as_float(uu[u] & 0xFFFF0000u);
          p += lo * wp[v2 * 8 + 2 * u] + hi * wp[v2 * 8 + 2 * u + 1];
        }
      }
      pre_s[tid] = p;
    }
    __syncthreads();

    // --- finish logits + softmax, one thread per batch row ---
    if (tid < kBM) {
      float lg[kC];
#pragma unroll
      for (int cc = 0; cc < kC; ++cc) {
        float sum = by_s[cc];
#pragma unroll
        for (int s = 0; s < 4; ++s) sum += pre_s[tid * 12 + cc * 4 + s];
        lg[cc] = sum;
      }
      float mx = fmaxf(lg[0], fmaxf(lg[1], lg[2]));
      float e0 = __expf(lg[0] - mx), e1 = __expf(lg[1] - mx), e2 = __expf(lg[2] - mx);
      float inv = 1.0f / (e0 + e1 + e2);
      size_t o = ((size_t)(b0 + tid) * kT + t) * kC;
      outP[o + 0] = e0 * inv; outP[o + 1] = e1 * inv; outP[o + 2] = e2 * inv;
    }
    __syncthreads();
  }
}

extern "C" void kernel_launch(void* const* d_in, const int* in_sizes, int n_in,
                              void* d_out, int out_size, void* d_ws, size_t ws_size,
                              hipStream_t stream) {
  const float* x  = (const float*)d_in[0];
  const float* Wf = (const float*)d_in[1];
  const float* Wi = (const float*)d_in[2];
  const float* Wc = (const float*)d_in[3];
  const float* Wo = (const float*)d_in[4];
  const float* Wy = (const float*)d_in[5];
  const float* bf = (const float*)d_in[6];
  const float* bi = (const float*)d_in[7];
  const float* bc = (const float*)d_in[8];
  const float* bo = (const float*)d_in[9];
  const float* by = (const float*)d_in[10];

  unsigned short* BX = (unsigned short*)d_ws;              // 9*32*512 bf16
  unsigned short* BH = BX + (size_t)kKTX * kNT * 512;      // 4*32*512 bf16

  float* outH = (float*)d_out;                             // [B,T,H]
  float* outP = outH + (size_t)kB * kT * kH;               // [B,T,C]

  int prepN = (kKTX + kKTH) * kNT * 512;                   // 212992 elements
  prep_weights<<<(prepN + 255) / 256, 256, 0, stream>>>(Wf, Wi, Wc, Wo, BX, BH);
  lstm_persistent<<<kB / kBM, kThreads, 0, stream>>>(x, Wy, bf, bi, bc, bo, by,
                                                     BX, BH, outH, outP);
}
